// BiMamba_21509196218760
// MI455X (gfx1250) — matrix operand order repaired and checked by reading, not compiled, not run on hardware
//
#include <hip/hip_runtime.h>
#include <hip/hip_bf16.h>
#include <math.h>

// ---- model dims (match reference) ----
#define D_MODEL   768
#define D_STATE   16
#define HEADDIM   64
#define D_CONV    4
#define D_INNER   1536            // 2*768
#define NHEADS    24              // 1536/64
#define CONV_DIM  1568            // 1536 + 2*16
#define D_IN_PROJ 3128            // 2*1536 + 2*16 + 24
#define BATCH     2
#define SEQ       2048            // power of two: flip uses shift/mask
#define SEQ_SHIFT 11
#define NROWS     (BATCH * SEQ)   // 4096
#define LDZ       3136            // D_IN_PROJ padded to multiple of 64
#define EPS       1e-5f

typedef __attribute__((ext_vector_type(16))) __bf16   v16bf;
typedef __attribute__((ext_vector_type(8)))  __bf16   v8bf;
typedef __attribute__((ext_vector_type(8)))  float    v8f;

// Round-to-nearest-even f32->bf16 pair packed into one dword via v_perm_b32.
__device__ __forceinline__ unsigned pack2_bf16(float x, float y) {
  unsigned ux = __builtin_bit_cast(unsigned, x);
  unsigned uy = __builtin_bit_cast(unsigned, y);
  ux += 0x7fffu + ((ux >> 16) & 1u);
  uy += 0x7fffu + ((uy >> 16) & 1u);
  return __builtin_amdgcn_perm(uy, ux, 0x07060302u);
}

__device__ __forceinline__ __bf16 f2bf(float f) {
  unsigned u = __builtin_bit_cast(unsigned, f);
  u += 0x7fffu + ((u >> 16) & 1u);
  unsigned short h = (unsigned short)(u >> 16);
  return __builtin_bit_cast(__bf16, h);
}

__device__ __forceinline__ float silu(float v) {
  return v / (1.f + __expf(-v));
}

__device__ __forceinline__ v16bf cat8(v8bf lo, v8bf hi) {
  return __builtin_shufflevector(lo, hi, 0, 1, 2, 3, 4, 5, 6, 7,
                                 8, 9, 10, 11, 12, 13, 14, 15);
}

// -----------------------------------------------------------------------------
// Elementwise f32 -> bf16 (2 elements/thread). n2 = element_count / 2.
// -----------------------------------------------------------------------------
__global__ __launch_bounds__(256) void cvt_f32_bf16(
    const float* __restrict__ in, __bf16* __restrict__ out, long n2)
{
  long i = (long)blockIdx.x * 256 + threadIdx.x;
  if (i >= n2) return;
  float2 v = ((const float2*)in)[i];
  ((unsigned*)out)[i] = pack2_bf16(v.x, v.y);
}

// -----------------------------------------------------------------------------
// C[M,N] = A[M,K] @ W[N,K]^T (+bias); A and W are bf16, accumulate f32 with
// v_wmma_f32_16x16x32_bf16. Block = 256 threads = 8 waves; each wave owns a
// 32x64 strip (2 A fragments x 4 B fragments = 8 WMMAs per K-step, B reused
// across both M-tiles). Block tile = 256(M) x 64(N).
// Out-of-range N columns: clamp the W row pointer only -- in WMMA, B column n
// is sourced solely from lane n, so garbage lands only in never-stored
// columns. Store is guarded. M % 256 == 0, K % 32 == 0 required.
// flipA!=0 reads A rows time-flipped within each SEQ-length sequence.
// STORE_BF16: 0 -> C is float*, 1 -> C is __bf16*.
// -----------------------------------------------------------------------------
template <int STORE_BF16>
__global__ __launch_bounds__(256) void gemm_bf16_wmma(
    const __bf16* __restrict__ A, int lda,
    const __bf16* __restrict__ W, int ldw,
    void* __restrict__ Cp, int ldc,
    int M, int N, int K,
    const float* __restrict__ bias,
    int flipA)
{
  const int lane = threadIdx.x & 31;
  const int wave = threadIdx.x >> 5;
  const int m0 = blockIdx.y * 256 + wave * 32;
  const int n0 = blockIdx.x * 64;
  if (m0 >= M) return;                 // wave-uniform

  const int half = lane >> 4;          // 0: lanes 0-15, 1: lanes 16-31
  const int l15  = lane & 15;

  // Two A fragment rows (16-bit A 16x32 layout: lane%16 = M row).
  const __bf16* Arow[2];
#pragma unroll
  for (int s = 0; s < 2; ++s) {
    int am = m0 + s * 16 + l15;
    if (flipA) {
      int b = am >> SEQ_SHIFT;
      int l = am & (SEQ - 1);
      am = (b << SEQ_SHIFT) + (SEQ - 1 - l);
    }
    Arow[s] = A + (long)am * lda;
  }

  // Four B fragment rows (W row n supplies B column n); clamp only.
  const __bf16* Wrow[4];
#pragma unroll
  for (int t = 0; t < 4; ++t) {
    int bn = n0 + t * 16 + l15;
    Wrow[t] = W + (long)((bn < N) ? bn : (N - 1)) * ldw;
  }

  v8f acc[2][4] = {{{}, {}, {}, {}}, {{}, {}, {}, {}}};

  for (int kk = 0; kk < K; kk += 32) {
    // Prefetch next K-chunk (global_prefetch_b8; speculative, OOB dropped).
    __builtin_prefetch(Arow[0] + kk + 32, 0, 1);
    __builtin_prefetch(Wrow[0] + kk + 32, 0, 1);
    __builtin_prefetch(Wrow[2] + kk + 32, 0, 1);

    // A fragments: lanes 0-15 hold K = kk+[0..7] and kk+16+[0..7];
    // lanes 16-31 the same +8. Two 16B spans -> 2x b128 each.
    v16bf af[2];
#pragma unroll
    for (int s = 0; s < 2; ++s) {
      const __bf16* ap = Arow[s] + kk + half * 8;
      af[s] = cat8(*(const v8bf*)ap, *(const v8bf*)(ap + 16));
    }

#pragma unroll
    for (int t = 0; t < 4; ++t) {
      // B 32x16 layout: same K order as A (lanes 0-15 K=kk+[0..7],kk+16+[0..7]; lanes 16-31 +8).
      const __bf16* wp = Wrow[t] + kk + half * 8;
      v16bf bfrag = cat8(*(const v8bf*)wp, *(const v8bf*)(wp + 16));
      acc[0][t] = __builtin_amdgcn_wmma_f32_16x16x32_bf16(
          false, af[0], false, bfrag, (short)0, acc[0][t], false, false);
      acc[1][t] = __builtin_amdgcn_wmma_f32_16x16x32_bf16(
          false, af[1], false, bfrag, (short)0, acc[1][t], false, false);
    }
  }

  // C/D 16x16 f32 layout: VGPR r holds M = r + 8*half, N = lane&15.
#pragma unroll
  for (int s = 0; s < 2; ++s) {
    const int mb = m0 + s * 16 + (half << 3);
#pragma unroll
    for (int t = 0; t < 4; ++t) {
      int cn = n0 + t * 16 + l15;
      if (cn < N) {
        float badd = bias ? bias[cn] : 0.f;
        if (STORE_BF16) {
          __bf16* Cb = (__bf16*)Cp;
#pragma unroll
          for (int r = 0; r < 8; ++r)
            Cb[(long)(mb + r) * ldc + cn] = f2bf(acc[s][t][r] + badd);
        } else {
          float* Cf = (float*)Cp;
#pragma unroll
          for (int r = 0; r < 8; ++r)
            Cf[(long)(mb + r) * ldc + cn] = acc[s][t][r] + badd;
        }
      }
    }
  }
}

// -----------------------------------------------------------------------------
// Depthwise causal conv1d (width 4) + bias + SiLU over the xBC slice.
// zx points at column D_INNER of the (NROWS, LDZ) zxbcdt matrix.
// -----------------------------------------------------------------------------
__global__ __launch_bounds__(256) void conv_silu(
    const float* __restrict__ zx, int ldz,
    const float* __restrict__ cw,     // (CONV_DIM, 4)
    const float* __restrict__ cb,     // (CONV_DIM,)
    float* __restrict__ out)          // (NROWS, CONV_DIM)
{
  long idx = (long)blockIdx.x * blockDim.x + threadIdx.x;
  const long total = (long)NROWS * CONV_DIM;
  if (idx >= total) return;
  int c  = (int)(idx % CONV_DIM);
  long m = idx / CONV_DIM;
  int l  = (int)(m & (SEQ - 1));
  long b = m >> SEQ_SHIFT;
  float acc = cb[c];
#pragma unroll
  for (int j = 0; j < D_CONV; ++j) {
    int ls = l - (D_CONV - 1) + j;
    float v = (ls >= 0) ? zx[((b << SEQ_SHIFT) + ls) * (long)ldz + c] : 0.f;
    acc += v * cw[c * D_CONV + j];
  }
  out[m * CONV_DIM + c] = silu(acc);
}

// -----------------------------------------------------------------------------
// Sequential SSM scan. One block per (batch, head). Each thread owns 4 state
// cells: p = tid>>2 (0..63), n = (tid&3)*4 .. +3. y reduced over n via two
// shfl_xor steps inside the wave32 (4 lanes per p).
// dtcol points at column D_INNER+CONV_DIM of zxbcdt (stride ldz).
// -----------------------------------------------------------------------------
__global__ __launch_bounds__(256) void ssm_scan(
    const float* __restrict__ dtcol, int ldz,
    const float* __restrict__ xBC,              // (NROWS, CONV_DIM)
    const float* __restrict__ dt_bias,          // (NHEADS,)
    const float* __restrict__ A_log,            // (NHEADS,)
    const float* __restrict__ Dp,               // (NHEADS,)
    float* __restrict__ y)                      // (NROWS, D_INNER)
{
  const int b = blockIdx.x / NHEADS;
  const int h = blockIdx.x % NHEADS;
  const int tid = threadIdx.x;
  const int p  = tid >> 2;
  const int nb = (tid & 3) << 2;

  float h0 = 0.f, h1 = 0.f, h2 = 0.f, h3 = 0.f;
  const float nA  = -__expf(A_log[h]);
  const float dtb = dt_bias[h];
  const float dp  = Dp[h];

  for (int l = 0; l < SEQ; ++l) {
    const long row = (long)b * SEQ + l;
    float dtr = dtcol[row * ldz + h] + dtb;
    float dt  = (dtr > 20.f) ? dtr : log1pf(__expf(dtr));  // softplus
    float dA  = __expf(nA * dt);

    const float* __restrict__ xr = xBC + row * CONV_DIM;
    float xv = xr[h * HEADDIM + p];
    const float* __restrict__ Bv = xr + D_INNER;
    const float* __restrict__ Cv = xr + D_INNER + D_STATE;

    float dtx = dt * xv;
    h0 = h0 * dA + dtx * Bv[nb + 0];
    h1 = h1 * dA + dtx * Bv[nb + 1];
    h2 = h2 * dA + dtx * Bv[nb + 2];
    h3 = h3 * dA + dtx * Bv[nb + 3];

    float part = h0 * Cv[nb + 0] + h1 * Cv[nb + 1]
               + h2 * Cv[nb + 2] + h3 * Cv[nb + 3];
    part += __shfl_xor(part, 1, 32);
    part += __shfl_xor(part, 2, 32);
    if ((tid & 3) == 0)
      y[row * D_INNER + h * HEADDIM + p] = part + dp * xv;
  }
}

// -----------------------------------------------------------------------------
// yh = bf16( rmsnorm(y * silu(z)) * norm_w ). One block per row.
// z is column 0..D_INNER-1 of zxbcdt (stride ldz).
// -----------------------------------------------------------------------------
__global__ __launch_bounds__(256) void gated_rmsnorm(
    const float* __restrict__ y,
    const float* __restrict__ zx, int ldz,
    const float* __restrict__ norm_w,
    __bf16* __restrict__ yh)
{
  __shared__ float vals[D_INNER];
  __shared__ float red[8];
  const long m = blockIdx.x;

  float ss = 0.f;
  for (int c = threadIdx.x; c < D_INNER; c += 256) {
    float z = zx[m * ldz + c];
    float v = y[m * D_INNER + c] * silu(z);
    vals[c] = v;
    ss += v * v;
  }
  ss += __shfl_xor(ss, 1, 32);
  ss += __shfl_xor(ss, 2, 32);
  ss += __shfl_xor(ss, 4, 32);
  ss += __shfl_xor(ss, 8, 32);
  ss += __shfl_xor(ss, 16, 32);
  if ((threadIdx.x & 31) == 0) red[threadIdx.x >> 5] = ss;
  __syncthreads();
  if (threadIdx.x < 32) {
    float t = (threadIdx.x < 8) ? red[threadIdx.x] : 0.f;
    t += __shfl_xor(t, 1, 32);
    t += __shfl_xor(t, 2, 32);
    t += __shfl_xor(t, 4, 32);
    if (threadIdx.x == 0) red[0] = t;
  }
  __syncthreads();
  const float r = rsqrtf(red[0] / (float)D_INNER + EPS);
  for (int c = threadIdx.x; c < D_INNER; c += 256)
    yh[m * D_INNER + c] = f2bf(vals[c] * r * norm_w[c]);
}

// -----------------------------------------------------------------------------
// Host-side launch. Inputs (all f32, setup_inputs order):
//  0:x 1:proj_w 2:proj_b
//  3:f_in_w 4:f_conv_w 5:f_conv_b 6:f_dt_bias 7:f_A_log 8:f_Dp 9:f_norm_w 10:f_out_w
//  11..18: b_* (same order)
// -----------------------------------------------------------------------------
extern "C" void kernel_launch(void* const* d_in, const int* in_sizes, int n_in,
                              void* d_out, int out_size, void* d_ws, size_t ws_size,
                              hipStream_t stream) {
  const float* x      = (const float*)d_in[0];
  const float* proj_w = (const float*)d_in[1];
  const float* proj_b = (const float*)d_in[2];

  const float* in_w[2]    = { (const float*)d_in[3],  (const float*)d_in[11] };
  const float* conv_w[2]  = { (const float*)d_in[4],  (const float*)d_in[12] };
  const float* conv_b[2]  = { (const float*)d_in[5],  (const float*)d_in[13] };
  const float* dt_bias[2] = { (const float*)d_in[6],  (const float*)d_in[14] };
  const float* A_log[2]   = { (const float*)d_in[7],  (const float*)d_in[15] };
  const float* Dp[2]      = { (const float*)d_in[8],  (const float*)d_in[16] };
  const float* norm_w[2]  = { (const float*)d_in[9],  (const float*)d_in[17] };
  const float* out_w[2]   = { (const float*)d_in[10], (const float*)d_in[18] };

  float* out = (float*)d_out;

  // workspace carving (256B aligned chunks)
  char* wp = (char*)d_ws;
  auto carve = [&](size_t bytes) -> void* {
    void* r = (void*)wp;
    wp += (bytes + 255) & ~(size_t)255;
    return r;
  };
  float* zx[2]   = { (float*)carve((size_t)NROWS * LDZ * 4),
                     (float*)carve((size_t)NROWS * LDZ * 4) };
  float* xBC[2]  = { (float*)carve((size_t)NROWS * CONV_DIM * 4),
                     (float*)carve((size_t)NROWS * CONV_DIM * 4) };
  float* yb[2]   = { (float*)carve((size_t)NROWS * D_INNER * 4),
                     (float*)carve((size_t)NROWS * D_INNER * 4) };
  __bf16* ybh[2] = { (__bf16*)carve((size_t)NROWS * D_INNER * 2),
                     (__bf16*)carve((size_t)NROWS * D_INNER * 2) };
  __bf16* cat    = (__bf16*)carve((size_t)NROWS * 2 * D_MODEL * 2);
  __bf16* xh     = (__bf16*)carve((size_t)NROWS * D_MODEL * 2);
  __bf16* inwh[2]  = { (__bf16*)carve((size_t)D_IN_PROJ * D_MODEL * 2),
                       (__bf16*)carve((size_t)D_IN_PROJ * D_MODEL * 2) };
  __bf16* outwh[2] = { (__bf16*)carve((size_t)D_MODEL * D_INNER * 2),
                       (__bf16*)carve((size_t)D_MODEL * D_INNER * 2) };
  __bf16* projwh   = (__bf16*)carve((size_t)D_MODEL * 2 * D_MODEL * 2);

  const dim3 blk(256);
  auto cvt = [&](const float* src, __bf16* dst, long n) {
    long n2 = n / 2;
    cvt_f32_bf16<<<dim3((unsigned)((n2 + 255) / 256)), blk, 0, stream>>>(src, dst, n2);
  };

  // 0) one-time (per call) bf16 conversions of GEMM operands
  cvt(x, xh, (long)NROWS * D_MODEL);
  cvt(in_w[0], inwh[0], (long)D_IN_PROJ * D_MODEL);
  cvt(in_w[1], inwh[1], (long)D_IN_PROJ * D_MODEL);
  cvt(out_w[0], outwh[0], (long)D_MODEL * D_INNER);
  cvt(out_w[1], outwh[1], (long)D_MODEL * D_INNER);
  cvt(proj_w, projwh, (long)D_MODEL * 2 * D_MODEL);

  // 1) in_proj GEMMs: zxbcdt = x @ in_w^T  (bwd reads x time-flipped)
  {
    dim3 grid((D_IN_PROJ + 63) / 64, NROWS / 256);
    for (int d = 0; d < 2; ++d)
      gemm_bf16_wmma<0><<<grid, blk, 0, stream>>>(
          xh, D_MODEL, inwh[d], D_MODEL, zx[d], LDZ,
          NROWS, D_IN_PROJ, D_MODEL, nullptr, /*flipA=*/d);
  }

  // 2) depthwise conv + SiLU on the xBC slice
  {
    long total = (long)NROWS * CONV_DIM;
    dim3 grid((unsigned)((total + 255) / 256));
    for (int d = 0; d < 2; ++d)
      conv_silu<<<grid, blk, 0, stream>>>(
          zx[d] + D_INNER, LDZ, conv_w[d], conv_b[d], xBC[d]);
  }

  // 3) sequential SSM scan, one block per (batch, head)
  {
    dim3 grid(BATCH * NHEADS);
    for (int d = 0; d < 2; ++d)
      ssm_scan<<<grid, blk, 0, stream>>>(
          zx[d] + (D_INNER + CONV_DIM), LDZ, xBC[d],
          dt_bias[d], A_log[d], Dp[d], yb[d]);
  }

  // 4) gated RMSNorm -> bf16
  {
    dim3 grid(NROWS);
    for (int d = 0; d < 2; ++d)
      gated_rmsnorm<<<grid, blk, 0, stream>>>(yb[d], zx[d], LDZ, norm_w[d], ybh[d]);
  }

  // 5) out_proj GEMMs into bf16 concat buffer; bwd writes time-flipped rows by
  //    reading its A rows flipped (equivalent to flipping the output).
  {
    dim3 grid(D_MODEL / 64, NROWS / 256);
    gemm_bf16_wmma<1><<<grid, blk, 0, stream>>>(
        ybh[0], D_INNER, outwh[0], D_INNER, cat, 2 * D_MODEL,
        NROWS, D_MODEL, D_INNER, nullptr, /*flipA=*/0);
    gemm_bf16_wmma<1><<<grid, blk, 0, stream>>>(
        ybh[1], D_INNER, outwh[1], D_INNER, cat + D_MODEL, 2 * D_MODEL,
        NROWS, D_MODEL, D_INNER, nullptr, /*flipA=*/1);
  }

  // 6) final projection: out = cat @ proj_w^T + proj_b
  {
    dim3 grid(D_MODEL / 64, NROWS / 256);
    gemm_bf16_wmma<0><<<grid, blk, 0, stream>>>(
        cat, 2 * D_MODEL, projwh, 2 * D_MODEL, out, D_MODEL,
        NROWS, D_MODEL, 2 * D_MODEL, proj_b, /*flipA=*/0);
  }
}
